// MESH_NET_34239479283955
// MI455X (gfx1250) — matrix operand order repaired and checked by reading, not compiled, not run on hardware
//
#include <hip/hip_runtime.h>
#include <hip/hip_bf16.h>

typedef __attribute__((ext_vector_type(16))) _Float16 v16h;
typedef __attribute__((ext_vector_type(8)))  _Float16 v8h;
typedef __attribute__((ext_vector_type(8)))  float    v8f;

#define B_    8
#define NPER  16000
#define DEGE  16
#define NTOT  (B_ * NPER)          // 128000
#define ETOT  (NTOT * DEGE)        // 2048000
#define HD    32
#define K1_   9600
#define K2_   5760
#define K3_   3456

// ---------------- utility kernels ----------------
__global__ void k_zero_f(float* p, long n) {
    long i = blockIdx.x * (long)blockDim.x + threadIdx.x;
    if (i < n) p[i] = 0.f;
}
__global__ void k_fill_f(float* p, float v, long n) {
    long i = blockIdx.x * (long)blockDim.x + threadIdx.x;
    if (i < n) p[i] = v;
}
__global__ void k_fill_i(int* p, int v, long n) {
    long i = blockIdx.x * (long)blockDim.x + threadIdx.x;
    if (i < n) p[i] = v;
}
__global__ void k_init_edges(const int* __restrict__ ein,
                             int* __restrict__ src, int* __restrict__ dst,
                             int* __restrict__ msk) {
    int e = blockIdx.x * blockDim.x + threadIdx.x;
    if (e < ETOT) { src[e] = ein[e]; dst[e] = ein[ETOT + e]; msk[e] = 1; }
}
__global__ void k_tohalf(const float* __restrict__ a, _Float16* __restrict__ o, long n) {
    long i = blockIdx.x * (long)blockDim.x + threadIdx.x;
    if (i < n) o[i] = (_Float16)a[i];
}
// pack W[32,32] f32 row-major into exact WMMA B-fragment layout (f16):
// wpack[(t*32 + lane)*16 + j] = W[(8*(lane>>4) + (j&7) + 16*(j>>3))*32 + t*16 + (lane&15)]
__global__ void k_packW(const float* __restrict__ W, _Float16* __restrict__ wpack) {
    int t = threadIdx.x;          // 64 threads: t>>5 = column-half, t&31 = lane
    if (t >= 64) return;
    int th = t >> 5, lane = t & 31;
    int ncol = lane & 15, hs = lane >> 4;
#pragma unroll
    for (int j = 0; j < 16; ++j)
        wpack[((size_t)th * 32 + lane) * 16 + j] =
            (_Float16)W[(hs * 8 + (j & 7) + (j >> 3) * 16) * HD + th * 16 + ncol];
}

// ---------------- GCN pieces ----------------
__global__ void k_deg(const int* __restrict__ dst, const int* __restrict__ msk,
                      float* __restrict__ deg) {
    int e = blockIdx.x * blockDim.x + threadIdx.x;
    if (e < ETOT && msk[e]) atomicAdd(&deg[dst[e]], 1.0f);
}
__global__ void k_dinv(const float* __restrict__ deg, float* __restrict__ dinv, int n) {
    int i = blockIdx.x * blockDim.x + threadIdx.x;
    if (i < n) dinv[i] = rsqrtf(deg[i]);
}
// stage-1 GEMM: x[n,3] @ W[3,32]
__global__ void k_gemm3(const float* __restrict__ x, const float* __restrict__ W,
                        float* __restrict__ xw, int n) {
    long t = blockIdx.x * (long)blockDim.x + threadIdx.x;
    if (t >= (long)n * HD) return;
    int i = (int)(t >> 5), c = (int)(t & 31);
    xw[t] = x[i*3+0]*W[0*HD+c] + x[i*3+1]*W[1*HD+c] + x[i*3+2]*W[2*HD+c];
}
// WMMA GEMM: A[n,32] f16 row-major @ packed-W -> C[n,32] f32.
// One wave per 16-node tile; 2x b128 loads for A, 2x b128 loads per B tile, 2 WMMAs.
__global__ void k_gemm_wmma(const _Float16* __restrict__ A,
                            const _Float16* __restrict__ wpack,
                            float* __restrict__ C, int n) {
    int lane = threadIdx.x & 31;
    int wave = threadIdx.x >> 5;
    int tile = blockIdx.x * (blockDim.x >> 5) + wave;
    int nodeBase = tile * 16;
    if (nodeBase >= n) return;                 // uniform per wave -> EXEC all-ones
    int m  = lane & 15;
    int hs = lane >> 4;                        // lane half select
    int ncol = lane & 15;
    // A 16x32 f16: lanes<16 hold K {0..7,16..23}; lanes>=16 hold K {8..15,24..31}
    const _Float16* arow = A + (size_t)(nodeBase + m) * HD;
    v8h alo = *(const v8h*)(arow + hs * 8);
    v8h ahi = *(const v8h*)(arow + 16 + hs * 8);
    v16h a = __builtin_shufflevector(alo, ahi, 0,1,2,3,4,5,6,7,8,9,10,11,12,13,14,15);
#pragma unroll
    for (int t = 0; t < 2; ++t) {              // two 16-column halves of the 32 outputs
        const _Float16* wb = wpack + ((size_t)t * 32 + lane) * 16;
        v8h blo = *(const v8h*)(wb);
        v8h bhi = *(const v8h*)(wb + 8);
        v16h b = __builtin_shufflevector(blo, bhi, 0,1,2,3,4,5,6,7,8,9,10,11,12,13,14,15);
        v8f c = {};
        c = __builtin_amdgcn_wmma_f32_16x16x32_f16(
                /*neg_a=*/false, a, /*neg_b=*/false, b,
                /*c_mod=*/(short)0, c, /*reuse_a=*/false, /*reuse_b=*/false);
        // D f32 16x16: VGPR r -> M = r + 8*hs, N = lane&15
        int mBase = nodeBase + 8 * hs;
#pragma unroll
        for (int r = 0; r < 8; ++r)
            C[(size_t)(mBase + r) * HD + t * 16 + ncol] = c[r];
    }
}
// message aggregation: agg[dst] += xw[src]*coef; 8 threads/edge, float4 per thread
__global__ void k_aggregate(const int* __restrict__ src, const int* __restrict__ dst,
                            const int* __restrict__ msk, const float* __restrict__ dinv,
                            const float* __restrict__ xw, float* __restrict__ agg) {
    long t = blockIdx.x * (long)blockDim.x + threadIdx.x;
    long e = t >> 3;
    if (e >= ETOT) return;
    if (!msk[e]) return;
    int s = src[e], d = dst[e];
    float coef = dinv[s] * dinv[d];
    int q = (int)(t & 7);
    const float4 v = *(reinterpret_cast<const float4*>(xw + (size_t)s * HD) + q);
    float* ab = agg + (size_t)d * HD + q * 4;
    atomicAdd(ab + 0, v.x * coef);
    atomicAdd(ab + 1, v.y * coef);
    atomicAdd(ab + 2, v.z * coef);
    atomicAdd(ab + 3, v.w * coef);
}
__global__ void k_gcn_final(const float* __restrict__ agg, const float* __restrict__ xw,
                            const float* __restrict__ dinv, const float* __restrict__ b,
                            float* __restrict__ h, int n) {
    long t = blockIdx.x * (long)blockDim.x + threadIdx.x;
    if (t >= (long)n * HD) return;
    int i = (int)(t >> 5), c = (int)(t & 31);
    float di = dinv[i];
    float v = agg[t] + xw[t] * (di * di) + b[c];
    h[t] = fmaxf(v, 0.f);
}

// ---------------- TopK pooling ----------------
__global__ void k_score(const float* __restrict__ h, const float* __restrict__ p,
                        float* __restrict__ score, int n) {
    int i = blockIdx.x * blockDim.x + threadIdx.x;
    if (i >= n) return;
    float pp = 0.f, dot = 0.f;
#pragma unroll
    for (int c = 0; c < HD; ++c) { float pc = p[c]; pp += pc * pc; dot += h[(size_t)i*HD+c] * pc; }
    score[i] = tanhf(dot * rsqrtf(pp));
}
// per-graph bitonic sort in LDS (descending score, ascending idx on ties)
__global__ void k_topk(const float* __restrict__ score, int* __restrict__ perm,
                       int nper, int kkeep, int PAD) {
    extern __shared__ char smem[];
    float* key = (float*)smem;
    int*   idx = (int*)(smem + (size_t)PAD * sizeof(float));
    int g = blockIdx.x, tid = threadIdx.x, bd = blockDim.x;
    for (int i = tid; i < PAD; i += bd) {
        if (i < nper) { key[i] = score[g * nper + i]; idx[i] = i; }
        else          { key[i] = -1e30f;              idx[i] = 0x7FFFFFFF; }
    }
    __syncthreads();
    for (int kk = 2; kk <= PAD; kk <<= 1) {
        for (int j = kk >> 1; j > 0; j >>= 1) {
            for (int i = tid; i < PAD; i += bd) {
                int ixj = i ^ j;
                if (ixj > i) {
                    float ka = key[i], kb = key[ixj];
                    int   ia = idx[i], ib = idx[ixj];
                    bool aFirst = (ka > kb) || (ka == kb && ia < ib);  // desc order
                    bool desc   = ((i & kk) == 0);
                    if (desc ? !aFirst : aFirst) {
                        key[i] = kb; key[ixj] = ka; idx[i] = ib; idx[ixj] = ia;
                    }
                }
            }
            __syncthreads();
        }
    }
    for (int i = tid; i < kkeep; i += bd) perm[g * kkeep + i] = g * nper + idx[i];
}
__global__ void k_scatter_map(const int* __restrict__ perm, int* __restrict__ mapping, int ntot_new) {
    int i = blockIdx.x * blockDim.x + threadIdx.x;
    if (i < ntot_new) mapping[perm[i]] = i;
}
__global__ void k_gather(const float* __restrict__ h, const float* __restrict__ score,
                         const int* __restrict__ perm, float* __restrict__ xnew, int ntot_new) {
    long t = blockIdx.x * (long)blockDim.x + threadIdx.x;
    if (t >= (long)ntot_new * HD) return;
    int i = (int)(t >> 5), c = (int)(t & 31);
    int p = perm[i];
    xnew[t] = h[(size_t)p * HD + c] * score[p];
}
__global__ void k_relabel(int* __restrict__ src, int* __restrict__ dst, int* __restrict__ msk,
                          const int* __restrict__ mapping) {
    int e = blockIdx.x * blockDim.x + threadIdx.x;
    if (e >= ETOT) return;
    int ns = mapping[src[e]], nd = mapping[dst[e]];
    int nm = msk[e] & (ns >= 0) & (nd >= 0);
    src[e] = ns > 0 ? ns : 0;
    dst[e] = nd > 0 ? nd : 0;
    msk[e] = nm;
}

// ---------------- readout + MLP ----------------
// one block per graph; 8 row-segments per channel + LDS reduce
__global__ void k_readout_add(const float* __restrict__ x, float* __restrict__ acc, int kper) {
    __shared__ float smax[256], ssum[256];
    int g = blockIdx.x, t = threadIdx.x;
    int c = t & 31, seg = t >> 5;               // 8 segments
    float mx = -3.4e38f, sm = 0.f;
    const float* base = x + (size_t)g * kper * HD + c;
    for (int r = seg; r < kper; r += 8) { float v = base[(size_t)r * HD]; mx = fmaxf(mx, v); sm += v; }
    smax[t] = mx; ssum[t] = sm;
    __syncthreads();
    if (seg == 0) {
#pragma unroll
        for (int s2 = 1; s2 < 8; ++s2) { mx = fmaxf(mx, smax[s2 * 32 + c]); sm += ssum[s2 * 32 + c]; }
        acc[g * 64 + c]      += mx;
        acc[g * 64 + 32 + c] += sm / (float)kper;
    }
}
__global__ void k_mlp(const float* __restrict__ acc,
                      const float* __restrict__ lw1, const float* __restrict__ lb1,
                      const float* __restrict__ lw2, const float* __restrict__ lb2,
                      float* __restrict__ out) {
    __shared__ float z1[B_ * 16];
    int t = threadIdx.x;
    if (t < B_ * 16) {
        int g = t / 16, j = t % 16;
        float s = lb1[j];
        for (int c = 0; c < 64; ++c) s += acc[g * 64 + c] * lw1[c * 16 + j];
        z1[t] = fmaxf(s, 0.f);
    }
    __syncthreads();
    if (t < B_ * 8) {
        int g = t / 8, j = t % 8;
        float s = lb2[j];
        for (int c = 0; c < 16; ++c) s += z1[g * 16 + c] * lw2[c * 8 + j];
        out[g * 8 + j] = 1.f / (1.f + expf(-s));
    }
}

// ---------------- host orchestration ----------------
static inline unsigned nblk(long n, int t) { return (unsigned)((n + t - 1) / t); }

extern "C" void kernel_launch(void* const* d_in, const int* in_sizes, int n_in,
                              void* d_out, int out_size, void* d_ws, size_t ws_size,
                              hipStream_t stream) {
    const float* x    = (const float*)d_in[0];
    const int*   ein  = (const int*)  d_in[1];   // edge_index [2,E]
    const float* W1   = (const float*)d_in[3];
    const float* b1   = (const float*)d_in[4];
    const float* p1   = (const float*)d_in[5];
    const float* W2   = (const float*)d_in[6];
    const float* b2   = (const float*)d_in[7];
    const float* p2   = (const float*)d_in[8];
    const float* W3   = (const float*)d_in[9];
    const float* b3   = (const float*)d_in[10];
    const float* p3   = (const float*)d_in[11];
    const float* lw1  = (const float*)d_in[12];
    const float* lb1  = (const float*)d_in[13];
    const float* lw2  = (const float*)d_in[14];
    const float* lb2  = (const float*)d_in[15];
    float* out = (float*)d_out;

    char* ws = (char*)d_ws;
    size_t off = 0;
    auto alloc = [&](size_t bytes) -> void* {
        void* p = ws + off;
        off = (off + bytes + 255) & ~(size_t)255;
        return p;
    };
    float*     hA    = (float*)    alloc((size_t)NTOT * HD * 4);
    float*     hB    = (float*)    alloc((size_t)NTOT * HD * 4);
    float*     xw    = (float*)    alloc((size_t)NTOT * HD * 4);
    float*     agg   = (float*)    alloc((size_t)NTOT * HD * 4);
    _Float16*  hhalf = (_Float16*) alloc((size_t)NTOT * HD * 2);
    _Float16*  wpack = (_Float16*) alloc((size_t)2 * 32 * 16 * 2);
    float*     deg   = (float*)    alloc((size_t)NTOT * 4);
    float*     dinv  = (float*)    alloc((size_t)NTOT * 4);
    float*     score = (float*)    alloc((size_t)NTOT * 4);
    int*       perm  = (int*)      alloc((size_t)B_ * K1_ * 4);
    int*       mapng = (int*)      alloc((size_t)NTOT * 4);
    int*       srcb  = (int*)      alloc((size_t)ETOT * 4);
    int*       dstb  = (int*)      alloc((size_t)ETOT * 4);
    int*       mskb  = (int*)      alloc((size_t)ETOT * 4);
    float*     acc   = (float*)    alloc((size_t)B_ * 64 * 4);
    (void)ws_size; (void)in_sizes; (void)n_in; (void)out_size;

    const int T = 256;
    k_init_edges<<<nblk(ETOT, T), T, 0, stream>>>(ein, srcb, dstb, mskb);
    k_zero_f<<<1, 256, 0, stream>>>(acc, B_ * 64);

    // per-stage parameters
    const int   nOld[3]  = { NTOT, B_ * K1_, B_ * K2_ };
    const int   nperS[3] = { NPER, K1_, K2_ };
    const int   kS[3]    = { K1_, K2_, K3_ };
    const int   padS[3]  = { 16384, 16384, 8192 };
    const float* Ws[3]   = { W1, W2, W3 };
    const float* bs[3]   = { b1, b2, b3 };
    const float* ps[3]   = { p1, p2, p3 };

    for (int s = 0; s < 3; ++s) {
        const int n = nOld[s];
        // ---- GCNConv: xw = h @ W ----
        if (s == 0) {
            k_gemm3<<<nblk((long)n * HD, T), T, 0, stream>>>(x, Ws[0], xw, n);
        } else {
            k_tohalf<<<nblk((long)n * HD, T), T, 0, stream>>>(hB, hhalf, (long)n * HD);
            k_packW<<<1, 64, 0, stream>>>(Ws[s], wpack);
            // 8 waves / block, one 16-node tile per wave, 2 WMMAs per tile
            k_gemm_wmma<<<nblk(n / 16, 8), 256, 0, stream>>>(hhalf, wpack, xw, n);
        }
        k_fill_f<<<nblk(n, T), T, 0, stream>>>(deg, 1.0f, n);
        k_deg<<<nblk(ETOT, T), T, 0, stream>>>(dstb, mskb, deg);
        k_dinv<<<nblk(n, T), T, 0, stream>>>(deg, dinv, n);
        k_zero_f<<<nblk((long)n * HD, T), T, 0, stream>>>(agg, (long)n * HD);
        k_aggregate<<<nblk((long)ETOT * 8, T), T, 0, stream>>>(srcb, dstb, mskb, dinv, xw, agg);
        k_gcn_final<<<nblk((long)n * HD, T), T, 0, stream>>>(agg, xw, dinv, bs[s], hA, n);
        // ---- TopK pool ----
        k_score<<<nblk(n, T), T, 0, stream>>>(hA, ps[s], score, n);
        k_topk<<<B_, 1024, (size_t)padS[s] * 8, stream>>>(score, perm, nperS[s], kS[s], padS[s]);
        const int nNew = B_ * kS[s];
        k_fill_i<<<nblk(n, T), T, 0, stream>>>(mapng, -1, n);
        k_scatter_map<<<nblk(nNew, T), T, 0, stream>>>(perm, mapng, nNew);
        k_gather<<<nblk((long)nNew * HD, T), T, 0, stream>>>(hA, score, perm, hB, nNew);
        k_relabel<<<nblk(ETOT, T), T, 0, stream>>>(srcb, dstb, mskb, mapng);
        // ---- readout (max ‖ mean), accumulated across stages ----
        k_readout_add<<<B_, 256, 0, stream>>>(hB, acc, kS[s]);
    }
    k_mlp<<<1, 128, 0, stream>>>(acc, lw1, lb1, lw2, lb2, out);
}
